// MSA_36739150250155
// MI455X (gfx1250) — matrix operand order repaired and checked by reading, not compiled, not run on hardware
//
#include <hip/hip_runtime.h>
#include <hip/hip_bf16.h>

typedef __attribute__((ext_vector_type(16))) _Float16 v16h;
typedef __attribute__((ext_vector_type(8)))  _Float16 half8;
typedef __attribute__((ext_vector_type(8)))  float    v8f;

#define WMMA_F16(a, b, c) \
  __builtin_amdgcn_wmma_f32_16x16x32_f16(false, (a), false, (b), (short)0, (c), false, false)

__device__ __forceinline__ v16h pack2(half8 lo, half8 hi) {
  v16h r;
#pragma unroll
  for (int i = 0; i < 8; i++) { r[i] = lo[i]; r[i + 8] = hi[i]; }
  return r;
}

// ---------------------------------------------------------------- converts
__global__ __launch_bounds__(256) void f32_to_f16_k(const float* __restrict__ in,
                                                    _Float16* __restrict__ out, int n) {
  int i = blockIdx.x * 256 + threadIdx.x;
  if (i < n) out[i] = (_Float16)in[i];
}

// (B,C,HW) f32 -> (B,HW,C) f16, 64x64 tiles through LDS
__global__ __launch_bounds__(256)
void transpose_cvt_k(const float* __restrict__ in, _Float16* __restrict__ out) {
  const int C = 512, HW = 4096;
  __shared__ __align__(16) _Float16 tile[64][72];  // [s_local][c_local]
  const int s0 = blockIdx.x * 64, c0 = blockIdx.y * 64, b = blockIdx.z;
  const int t = threadIdx.x;
  const int row = t >> 2, seg = (t & 3) * 16;

  const float* ip = in + ((size_t)b * C + c0 + row) * HW + s0 + seg;
#pragma unroll
  for (int jj = 0; jj < 4; jj++) {
    float4 f = ((const float4*)ip)[jj];
    tile[seg + jj * 4 + 0][row] = (_Float16)f.x;
    tile[seg + jj * 4 + 1][row] = (_Float16)f.y;
    tile[seg + jj * 4 + 2][row] = (_Float16)f.z;
    tile[seg + jj * 4 + 3][row] = (_Float16)f.w;
  }
  __syncthreads();
  _Float16* op = out + ((size_t)b * HW + s0 + row) * C + c0 + seg;
  *(half8*)op       = *(const half8*)&tile[row][seg];
  *(half8*)(op + 8) = *(const half8*)&tile[row][seg + 8];
}

// ---------------------------------------------------------------- pointwise GEMM (no LDS)
// Y[m,n] = sum_k W[m,k] * Xt[n,k] + bias[m]; M=512, K=512, N=4096 per batch.
// Block: 8 waves, each wave: 2 m-tiles x 4 n-tiles -> block covers 128x128.
// OUTMODE 0: f16 (C,HW) [q,k]; 1: f16 (HW,C) b128 stores [v]; 2: f32 (C,HW)+resid [out]
template <int OUTMODE>
__global__ __launch_bounds__(256)
void gemm_pw(const _Float16* __restrict__ W, const float* __restrict__ bias,
             const _Float16* __restrict__ Xt, _Float16* __restrict__ Yh,
             float* __restrict__ Yf, const float* __restrict__ resid) {
  const int C = 512, HW = 4096;
  const int nblk = blockIdx.x * 128;
  const int mblk = blockIdx.y * 128;
  const int b = blockIdx.z;
  const int t = threadIdx.x, lane = t & 31, wave = t >> 5;
  const int l15 = lane & 15, g = lane >> 4;
  const int mt0 = (wave & 3) * 2;   // 2 m-tiles
  const int nt0 = (wave >> 2) * 4;  // 4 n-tiles

  const _Float16* wrow0 = W + (size_t)(mblk + (mt0 + 0) * 16 + l15) * C;
  const _Float16* wrow1 = W + (size_t)(mblk + (mt0 + 1) * 16 + l15) * C;
  const _Float16* xrow[4];
#pragma unroll
  for (int i = 0; i < 4; i++)
    xrow[i] = Xt + ((size_t)b * HW + nblk + (nt0 + i) * 16 + l15) * C + 16 * g;

  v8f acc[2][4];
#pragma unroll
  for (int mi = 0; mi < 2; mi++)
#pragma unroll
    for (int nt = 0; nt < 4; nt++) acc[mi][nt] = (v8f){};

  for (int k0 = 0; k0 < C; k0 += 32) {
    if (k0 + 32 < C) {
#pragma unroll
      for (int nt = 0; nt < 4; nt++) __builtin_prefetch(xrow[nt] + k0 + 32, 0, 1);
    }
    v16h a0 = pack2(*(const half8*)(wrow0 + k0 + 16 * g),
                    *(const half8*)(wrow0 + k0 + 16 * g + 8));
    v16h a1 = pack2(*(const half8*)(wrow1 + k0 + 16 * g),
                    *(const half8*)(wrow1 + k0 + 16 * g + 8));
#pragma unroll
    for (int nt = 0; nt < 4; nt++) {
      v16h bf = pack2(*(const half8*)(xrow[nt] + k0), *(const half8*)(xrow[nt] + k0 + 8));
      acc[0][nt] = WMMA_F16(a0, bf, acc[0][nt]);
      acc[1][nt] = WMMA_F16(a1, bf, acc[1][nt]);
    }
  }

#pragma unroll
  for (int mi = 0; mi < 2; mi++) {
    const int mbase = mblk + (mt0 + mi) * 16 + 8 * g;  // + r
#pragma unroll
    for (int nt = 0; nt < 4; nt++) {
      const int n = nblk + (nt0 + nt) * 16 + l15;
      if (OUTMODE == 1) {
        half8 hv;
#pragma unroll
        for (int r = 0; r < 8; r++) hv[r] = (_Float16)(acc[mi][nt][r] + bias[mbase + r]);
        *(half8*)(Yh + ((size_t)b * HW + n) * C + mbase) = hv;
      } else {
#pragma unroll
        for (int r = 0; r < 8; r++) {
          const int m = mbase + r;
          const size_t off = (size_t)b * C * HW + (size_t)m * HW + n;
          float vv = acc[mi][nt][r] + bias[m];
          if (OUTMODE == 2) Yf[off] = vv + resid[off];
          else              Yh[off] = (_Float16)vv;
        }
      }
    }
  }
}

// ---------------------------------------------------------------- attention
// per (b,head): as1 = q k^T (64x64, K=4096); softmax*8; DA_t = (A v) in NHWC
__global__ __launch_bounds__(256)
void attn_k(const _Float16* __restrict__ q, const _Float16* __restrict__ k,
            const _Float16* __restrict__ vt, _Float16* __restrict__ DAt,
            float* __restrict__ as1_out) {
  const int C = 512, CH = 64, HW = 4096;
  __shared__ __align__(16) float    s_as1[64][65];
  __shared__ __align__(16) _Float16 s_A[64][72];

  const int bh = blockIdx.x;
  const int b = bh >> 3, head = bh & 7;
  const int t = threadIdx.x, lane = t & 31, wave = t >> 5;
  const int l15 = lane & 15, g = lane >> 4;

  const _Float16* qb = q + (size_t)bh * CH * HW;
  const _Float16* kb = k + (size_t)bh * CH * HW;

  // ---- phase 1: as1 = q k^T (direct global -> WMMA streaming)
  const int mt = wave >> 1, ntb = (wave & 1) * 2;
  v8f acc0 = (v8f){}, acc1 = (v8f){};
  for (int i0 = 0; i0 < HW; i0 += 32) {
    const _Float16* qrow = qb + (size_t)(mt * 16 + l15) * HW + i0;
    v16h a = pack2(*(const half8*)(qrow + 16 * g), *(const half8*)(qrow + 16 * g + 8));
    const _Float16* k0p = kb + (size_t)((ntb + 0) * 16 + l15) * HW + i0 + 16 * g;
    v16h b0 = pack2(*(const half8*)k0p, *(const half8*)(k0p + 8));
    acc0 = WMMA_F16(a, b0, acc0);
    const _Float16* k1p = kb + (size_t)((ntb + 1) * 16 + l15) * HW + i0 + 16 * g;
    v16h b1 = pack2(*(const half8*)k1p, *(const half8*)(k1p + 8));
    acc1 = WMMA_F16(a, b1, acc1);
  }
#pragma unroll
  for (int r = 0; r < 8; r++) {
    s_as1[mt * 16 + r + 8 * g][(ntb + 0) * 16 + l15] = acc0[r];
    s_as1[mt * 16 + r + 8 * g][(ntb + 1) * 16 + l15] = acc1[r];
  }
  __syncthreads();

  // ---- phase 2: raw as1 -> second output
  float* ao = as1_out + (size_t)bh * 4096;
  for (int i = t; i < 4096; i += 256) ao[i] = s_as1[i >> 6][i & 63];

  // ---- phase 3: row softmax * sqrt(64)
  if (t < 64) {
    float mx = -1e30f;
    for (int j = 0; j < 64; j++) mx = fmaxf(mx, s_as1[t][j]);
    float sum = 0.f;
    for (int j = 0; j < 64; j++) sum += expf(s_as1[t][j] - mx);
    const float sc = 8.0f / sum;
    for (int j = 0; j < 64; j++) s_A[t][j] = (_Float16)(expf(s_as1[t][j] - mx) * sc);
  }
  __syncthreads();

  // ---- phase 4: DA = A v, B-frags direct from v_t (NHWC), b128 output stores
  const int mt4 = wave & 3, nts = wave >> 2;  // n-tiles: nts, nts+2, ...
  v16h afr[2];
#pragma unroll
  for (int ks = 0; ks < 2; ks++) {
    const _Float16* ap = &s_A[mt4 * 16 + l15][ks * 32];
    afr[ks] = pack2(*(const half8*)(ap + 16 * g), *(const half8*)(ap + 16 * g + 8));
  }

  for (int nt = nts; nt < 256; nt += 2) {
    const int col = nt * 16 + l15;
    const _Float16* vp = vt + ((size_t)b * HW + col) * C + head * 64 + 16 * g;
    v8f acc = (v8f){};
#pragma unroll
    for (int ks = 0; ks < 2; ks++) {
      v16h bf = pack2(*(const half8*)(vp + ks * 32), *(const half8*)(vp + ks * 32 + 8));
      acc = WMMA_F16(afr[ks], bf, acc);
    }
    half8 hv;
#pragma unroll
    for (int r = 0; r < 8; r++) hv[r] = (_Float16)acc[r];
    *(half8*)(DAt + ((size_t)b * HW + col) * C + head * 64 + mt4 * 16 + 8 * g) = hv;
  }
}

// ---------------------------------------------------------------- depthwise 7x7 + gelu + resid (NHWC)
__global__ __launch_bounds__(256)
void dwconv_gelu_k(const _Float16* __restrict__ DAt, const float* __restrict__ w,
                   const float* __restrict__ bias, _Float16* __restrict__ leakt) {
  const int C = 512, HW = 4096;
  const int idx = blockIdx.x * 256 + threadIdx.x;  // (b, s, c) NHWC order
  const int c = idx & 511;
  const int s = (idx >> 9) & 4095;
  const int b = idx >> 21;
  const int x = s & 63, y = (s >> 6) & 63;
  const float* wc = w + c * 49;
  const _Float16* pb = DAt + (size_t)b * HW * C + c;
  float acc = bias[c];
#pragma unroll
  for (int dy = -3; dy <= 3; dy++) {
    const int yy = y + dy;
    if (yy < 0 || yy > 63) continue;
#pragma unroll
    for (int dx = -3; dx <= 3; dx++) {
      const int xx = x + dx;
      if (xx < 0 || xx > 63) continue;
      acc += wc[(dy + 3) * 7 + (dx + 3)] * (float)pb[(size_t)(yy * 64 + xx) * C];
    }
  }
  const float gel = 0.5f * acc * (1.0f + erff(acc * 0.70710678118f));
  leakt[idx] = (_Float16)(gel + (float)DAt[idx]);
}

// ---------------------------------------------------------------- launcher
extern "C" void kernel_launch(void* const* d_in, const int* in_sizes, int n_in,
                              void* d_out, int out_size, void* d_ws, size_t ws_size,
                              hipStream_t stream) {
  (void)in_sizes; (void)n_in; (void)out_size; (void)ws_size;
  const int B = 8, C = 512, HW = 4096;
  const size_t NSTYLE = (size_t)B * C * HW;  // 16,777,216
  const size_t NW = (size_t)C * C;           // 262,144

  const float* style = (const float*)d_in[0];
  const float* wq = (const float*)d_in[1];  const float* bq = (const float*)d_in[2];
  const float* wk = (const float*)d_in[3];  const float* bk = (const float*)d_in[4];
  const float* wv = (const float*)d_in[5];  const float* bv = (const float*)d_in[6];
  const float* dww = (const float*)d_in[7]; const float* dwb = (const float*)d_in[8];
  const float* wo = (const float*)d_in[9];  const float* bo = (const float*)d_in[10];

  float* out_main = (float*)d_out;
  float* out_as1 = out_main + NSTYLE;

  // ws layout (f16) with reuse: DA_t overwrites style_t, leak_t overwrites q_h
  char* ws = (char*)d_ws;
  _Float16* style_t = (_Float16*)ws;     // (B,HW,C)
  _Float16* wq_h = style_t + NSTYLE;
  _Float16* wk_h = wq_h + NW;
  _Float16* wv_h = wk_h + NW;
  _Float16* wo_h = wv_h + NW;
  _Float16* q_h = wo_h + NW;             // (B,C,HW)
  _Float16* k_h = q_h + NSTYLE;          // (B,C,HW)
  _Float16* v_t = k_h + NSTYLE;          // (B,HW,C)
  _Float16* DA_t = style_t;              // (B,HW,C), reuse
  _Float16* leak_t = q_h;                // (B,HW,C), reuse

  // 1) converts
  transpose_cvt_k<<<dim3(HW / 64, C / 64, B), 256, 0, stream>>>(style, style_t);
  f32_to_f16_k<<<(int)(NW / 256), 256, 0, stream>>>(wq, wq_h, (int)NW);
  f32_to_f16_k<<<(int)(NW / 256), 256, 0, stream>>>(wk, wk_h, (int)NW);
  f32_to_f16_k<<<(int)(NW / 256), 256, 0, stream>>>(wv, wv_h, (int)NW);
  f32_to_f16_k<<<(int)(NW / 256), 256, 0, stream>>>(wo, wo_h, (int)NW);

  // 2) QKV GEMMs (LDS-free streaming WMMA)
  dim3 gg(HW / 128, C / 128, B);
  gemm_pw<0><<<gg, 256, 0, stream>>>(wq_h, bq, style_t, q_h, nullptr, nullptr);
  gemm_pw<0><<<gg, 256, 0, stream>>>(wk_h, bk, style_t, k_h, nullptr, nullptr);
  gemm_pw<1><<<gg, 256, 0, stream>>>(wv_h, bv, style_t, v_t, nullptr, nullptr);

  // 3) attention (raw as1 to out; DA_t over style_t)
  attn_k<<<64, 256, 0, stream>>>(q_h, k_h, v_t, DA_t, out_as1);

  // 4) depthwise 7x7 + exact GELU + residual(DA), NHWC
  dwconv_gelu_k<<<(int)(NSTYLE / 256), 256, 0, stream>>>(DA_t, dww, dwb, leak_t);

  // 5) output conv + bias + residual(style), f32 (C,HW) out
  gemm_pw<2><<<gg, 256, 0, stream>>>(wo_h, bo, leak_t, nullptr, out_main, style);
}
